// VlashQwen3Attention_16535624089976
// MI455X (gfx1250) — matrix operand order repaired and checked by reading, not compiled, not run on hardware
//
#include <hip/hip_runtime.h>
#include <stdint.h>

// Qwen3 attention block, MI455X (gfx1250, wave32, WMMA bf16 16x16x32,
// async global->LDS double buffering in the GEMMs).
#define S_LEN 2048
#define D_MODEL 4096
#define NH 32
#define NKV 8
#define HD 128

typedef __bf16 bf16_t;
typedef __attribute__((ext_vector_type(16))) __bf16 v16bf;
typedef __attribute__((ext_vector_type(8)))  float  v8f;

__device__ inline bf16_t f2bf(float f) {
  union { float f; uint32_t u; } x; x.f = f;
  uint32_t r = (x.u + 0x7FFFu + ((x.u >> 16) & 1u)) >> 16;   // RNE
  union { uint16_t u; bf16_t b; } y; y.u = (uint16_t)r;
  return y.b;
}

__device__ inline v8f vzero8() {
  v8f z;
#pragma unroll
  for (int i = 0; i < 8; ++i) z[i] = 0.0f;
  return z;
}

__device__ inline v8f wmma_bf16(v16bf a, v16bf b, v8f c) {
  // D = A(16x32 bf16) * B(32x16 bf16) + C(16x16 f32)
  return __builtin_amdgcn_wmma_f32_16x16x32_bf16(false, a, false, b, (short)0, c, false, false);
}

// ---- CDNA5 async global->LDS copy (ASYNCcnt-tracked, in-order completion) ----
__device__ inline void async_copy_b128(uint32_t lds_addr, const void* gaddr) {
  asm volatile("global_load_async_to_lds_b128 %0, %1, off"
               :: "v"(lds_addr), "v"(gaddr)
               : "memory");
}
__device__ inline void wait_async_le4() { asm volatile("s_wait_asynccnt 0x4" ::: "memory"); }
__device__ inline void wait_async_le0() { asm volatile("s_wait_asynccnt 0x0" ::: "memory"); }
__device__ inline uint32_t lds_addr_of(const void* p) {
  return (uint32_t)(uintptr_t)p;  // generic LDS addr: low 32 bits = group-segment offset
}

// A-fragment: 16x32 bf16, source stored row-major [M][K] with leading dim ld.
// lane L: row = L&15; elements 0..7 <- K[k0..k0+7], 8..15 <- K[k0+16..k0+23], k0 = (L<16?0:8)
__device__ inline v16bf frag_a(const bf16_t* base, int ld, int lane) {
  const int r  = lane & 15;
  const int k0 = (lane & 16) ? 8 : 0;
  const bf16_t* p = base + (size_t)r * ld + k0;
  union { v16bf f; uint4 u[2]; } x;
  x.u[0] = *(const uint4*)(p);
  x.u[1] = *(const uint4*)(p + 16);
  return x.f;
}

// B-fragment: 32x16 bf16, source stored as B^T row-major [N][K] with leading dim ld.
// lane L: col n = L&15; elements 0..7 <- K[k0..k0+7], 8..15 <- K[k0+16..k0+23], k0 = (L<16?0:8)
__device__ inline v16bf frag_b(const bf16_t* base, int ld, int lane) {
  const int n  = lane & 15;
  const int k0 = (lane & 16) ? 8 : 0;
  const bf16_t* p = base + (size_t)n * ld + k0;
  union { v16bf f; uint4 u[2]; } x;
  x.u[0] = *(const uint4*)(p);
  x.u[1] = *(const uint4*)(p + 16);
  return x.f;
}

__device__ inline float rmax16(float v) {
#pragma unroll
  for (int o = 1; o < 16; o <<= 1) v = fmaxf(v, __shfl_xor(v, o, 32));
  return v;
}
__device__ inline float rsum16(float v) {
#pragma unroll
  for (int o = 1; o < 16; o <<= 1) v += __shfl_xor(v, o, 32);
  return v;
}
__device__ inline float rsum32(float v) {
#pragma unroll
  for (int o = 1; o < 32; o <<= 1) v += __shfl_xor(v, o, 32);
  return v;
}

// ---------------------------------------------------------------- convert
__global__ void cvt_f32_bf16_kernel(const float* __restrict__ src,
                                    bf16_t* __restrict__ dst, int n) {
  int i = blockIdx.x * blockDim.x + threadIdx.x;
  if (i < n) dst[i] = f2bf(src[i]);
}

// ---------------------------------------------------------------- GEMM
// C[M][N] (f32) = A[M][K] (bf16 row-major) * W[N][K]^T (bf16 row-major)
// block tile 128x128, 256 threads = 8 waves, wave tile 64x32.
// Double-buffered LDS filled by global_load_async_to_lds_b128.
template <int K>
__global__ __launch_bounds__(256) void gemm_bf16_kernel(
    const bf16_t* __restrict__ A, const bf16_t* __restrict__ W,
    float* __restrict__ C, int M, int N) {
  __shared__ __align__(16) bf16_t As[2][128][32];
  __shared__ __align__(16) bf16_t Bs[2][128][32];
  const int tid  = threadIdx.x;
  const int lane = tid & 31;
  const int w    = tid >> 5;
  const int wm   = w >> 2;      // 0..1
  const int wn   = w & 3;       // 0..3
  const int m0   = blockIdx.y * 128;
  const int n0   = blockIdx.x * 128;

  // each thread copies 2x16B of A and 2x16B of B per stage
  const int row0 = tid >> 2;            // 0..63
  const int row1 = row0 + 64;           // 64..127
  const int col0 = (tid & 3) * 8;       // bf16 elements, 16B aligned
  const bf16_t* gA0 = A + (size_t)(m0 + row0) * K + col0;
  const bf16_t* gA1 = A + (size_t)(m0 + row1) * K + col0;
  const bf16_t* gB0 = W + (size_t)(n0 + row0) * K + col0;
  const bf16_t* gB1 = W + (size_t)(n0 + row1) * K + col0;
  const uint32_t lA0[2] = {lds_addr_of(&As[0][row0][col0]), lds_addr_of(&As[1][row0][col0])};
  const uint32_t lA1[2] = {lds_addr_of(&As[0][row1][col0]), lds_addr_of(&As[1][row1][col0])};
  const uint32_t lB0[2] = {lds_addr_of(&Bs[0][row0][col0]), lds_addr_of(&Bs[1][row0][col0])};
  const uint32_t lB1[2] = {lds_addr_of(&Bs[0][row1][col0]), lds_addr_of(&Bs[1][row1][col0])};

  auto issue = [&](int buf, int k0) {
    async_copy_b128(lA0[buf], gA0 + k0);
    async_copy_b128(lA1[buf], gA1 + k0);
    async_copy_b128(lB0[buf], gB0 + k0);
    async_copy_b128(lB1[buf], gB1 + k0);
  };

  v8f acc[4][2];
#pragma unroll
  for (int mt = 0; mt < 4; ++mt)
#pragma unroll
    for (int nt = 0; nt < 2; ++nt) acc[mt][nt] = vzero8();

  constexpr int KT = K / 32;
  issue(0, 0);
  for (int kt = 0; kt < KT; ++kt) {
    const int cur = kt & 1;
    if (kt + 1 < KT) {
      issue(cur ^ 1, (kt + 1) * 32);
      wait_async_le4();   // 8 outstanding -> <=4 means current stage landed
    } else {
      wait_async_le0();
    }
    __syncthreads();      // current stage visible to all waves

    v16bf af[4], bfg[2];
#pragma unroll
    for (int mt = 0; mt < 4; ++mt)
      af[mt] = frag_a(&As[cur][wm * 64 + mt * 16][0], 32, lane);
#pragma unroll
    for (int nt = 0; nt < 2; ++nt)
      bfg[nt] = frag_b(&Bs[cur][wn * 32 + nt * 16][0], 32, lane);
#pragma unroll
    for (int mt = 0; mt < 4; ++mt)
#pragma unroll
      for (int nt = 0; nt < 2; ++nt)
        acc[mt][nt] = wmma_bf16(af[mt], bfg[nt], acc[mt][nt]);

    __syncthreads();      // all reads of `cur` done before it is refilled at kt+1
  }

  const int half = lane >> 4;
  const int nn   = lane & 15;
#pragma unroll
  for (int mt = 0; mt < 4; ++mt)
#pragma unroll
    for (int nt = 0; nt < 2; ++nt)
#pragma unroll
      for (int r = 0; r < 8; ++r) {
        const int row = m0 + wm * 64 + mt * 16 + r + 8 * half;
        const int col = n0 + wn * 32 + nt * 16 + nn;
        C[(size_t)row * N + col] = acc[mt][nt][r];
      }
}

// ---------------------------------------------------------------- QK norm + RoPE + layout
// one wave per (s, head) row; q -> qb[NH][S][HD], k -> kb[NKV][S][HD],
// v -> vbt[NKV][HD][S] (transposed for the PV B-fragments), all bf16.
__global__ __launch_bounds__(256) void qkv_post_kernel(
    const float* __restrict__ qf, const float* __restrict__ kf,
    const float* __restrict__ vf, const float* __restrict__ qw,
    const float* __restrict__ kw, const float* __restrict__ cosf,
    const float* __restrict__ sinf, bf16_t* __restrict__ qb,
    bf16_t* __restrict__ kb, bf16_t* __restrict__ vbt) {
  const int wave = threadIdx.x >> 5;
  const int lane = threadIdx.x & 31;
  const long rowid = (long)blockIdx.x * 8 + wave;
  const long QROWS = (long)S_LEN * NH;
  const long KROWS = (long)S_LEN * NKV;
  const int dbase = lane * 4;

  if (rowid < QROWS + KROWS) {
    // q or k row: RMSNorm + RoPE
    const bool isQ = rowid < QROWS;
    const int s = isQ ? (int)(rowid >> 5) : (int)((rowid - QROWS) >> 3);
    const int h = isQ ? (int)(rowid & 31) : (int)((rowid - QROWS) & 7);
    const float* src = isQ ? (qf + (size_t)s * (NH * HD) + h * HD)
                           : (kf + (size_t)s * (NKV * HD) + h * HD);
    const float* nw = isQ ? qw : kw;
    union { float4 v; float a[4]; } x;
    x.v = *(const float4*)(src + dbase);
    float ss = x.a[0]*x.a[0] + x.a[1]*x.a[1] + x.a[2]*x.a[2] + x.a[3]*x.a[3];
    ss = rsum32(ss);
    const float inv = rsqrtf(ss * (1.0f / HD) + 1e-6f);
    float xn[4];
#pragma unroll
    for (int i = 0; i < 4; ++i) xn[i] = x.a[i] * inv * nw[dbase + i];
    // RoPE: lanes 0..15 hold d=0..63 (x1), lanes 16..31 hold d=64..127 (x2)
    bf16_t outv[4];
#pragma unroll
    for (int i = 0; i < 4; ++i) {
      const int idx = (dbase + i) & 63;
      const float c  = cosf[(size_t)s * HD + idx];
      const float sn = sinf[(size_t)s * HD + idx];
      const float other = __shfl_xor(xn[i], 16, 32);
      const float o = (lane & 16) ? (xn[i] * c + other * sn)
                                  : (xn[i] * c - other * sn);
      outv[i] = f2bf(o);
    }
    bf16_t* dst = isQ ? (qb + ((size_t)h * S_LEN + s) * HD + dbase)
                      : (kb + ((size_t)h * S_LEN + s) * HD + dbase);
#pragma unroll
    for (int i = 0; i < 4; ++i) dst[i] = outv[i];
  } else if (rowid < QROWS + 2 * KROWS) {
    // v row: just transpose + convert
    const long r2 = rowid - QROWS - KROWS;
    const int s = (int)(r2 >> 3);
    const int h = (int)(r2 & 7);
    union { float4 v; float a[4]; } x;
    x.v = *(const float4*)(vf + (size_t)s * (NKV * HD) + h * HD + dbase);
#pragma unroll
    for (int i = 0; i < 4; ++i)
      vbt[((size_t)h * HD + dbase + i) * S_LEN + s] = f2bf(x.a[i]);
  }
}

// ---------------------------------------------------------------- flash attention
// block = 8 independent waves; wave owns 16 query rows of one head.
__global__ __launch_bounds__(256) void flash_attn_kernel(
    const bf16_t* __restrict__ qb, const bf16_t* __restrict__ kb,
    const bf16_t* __restrict__ vbt, bf16_t* __restrict__ attn) {
  __shared__ __align__(16) bf16_t plds[8][16][32];
  const int w    = threadIdx.x >> 5;
  const int lane = threadIdx.x & 31;
  const int nqb  = S_LEN / 128;             // 16
  const int head = blockIdx.x / nqb;
  const int qblk = blockIdx.x % nqb;
  const int q0   = qblk * 128 + w * 16;
  const int kvh  = head >> 2;               // GQA groups = 4
  const int half = lane >> 4;
  const int nn   = lane & 15;
  const float scale = 0.08838834764831845f; // 128^-0.5

  v16bf qfrag[4];
#pragma unroll
  for (int c = 0; c < 4; ++c)
    qfrag[c] = frag_a(qb + ((size_t)head * S_LEN + q0) * HD + c * 32, HD, lane);

  v8f o[8];
#pragma unroll
  for (int nt = 0; nt < 8; ++nt) o[nt] = vzero8();
  float m[8], l[8];
#pragma unroll
  for (int r = 0; r < 8; ++r) { m[r] = -1e30f; l[r] = 0.0f; }

  const int nkb = (q0 + 16 + 31) >> 5;      // 32-key blocks (causal bound)
  for (int kb32 = 0; kb32 < nkb; ++kb32) {
    const int kbase = kb32 * 32;
    v8f s0 = vzero8(), s1 = vzero8();
#pragma unroll
    for (int c = 0; c < 4; ++c) {
      v16bf kf0 = frag_b(kb + ((size_t)kvh * S_LEN + kbase) * HD + c * 32, HD, lane);
      v16bf kf1 = frag_b(kb + ((size_t)kvh * S_LEN + kbase + 16) * HD + c * 32, HD, lane);
      s0 = wmma_bf16(qfrag[c], kf0, s0);
      s1 = wmma_bf16(qfrag[c], kf1, s1);
    }
    // online softmax update (rows live in 16-lane halves)
    float alpha[8];
#pragma unroll
    for (int r = 0; r < 8; ++r) {
      const int row = q0 + r + 8 * half;
      float a = s0[r] * scale + ((kbase + nn)      > row ? -1e9f : 0.0f);
      float b = s1[r] * scale + ((kbase + 16 + nn) > row ? -1e9f : 0.0f);
      float t = rmax16(fmaxf(a, b));
      const float mn = fmaxf(m[r], t);
      alpha[r] = __expf(m[r] - mn);
      const float p0 = __expf(a - mn);
      const float p1 = __expf(b - mn);
      l[r] = l[r] * alpha[r] + rsum16(p0 + p1);
      m[r] = mn;
      const int prow = r + 8 * half;
      plds[w][prow][nn]      = f2bf(p0);
      plds[w][prow][16 + nn] = f2bf(p1);
    }
#pragma unroll
    for (int nt = 0; nt < 8; ++nt)
#pragma unroll
      for (int r = 0; r < 8; ++r) o[nt][r] *= alpha[r];
    // P (C-layout) -> A-layout through per-wave LDS tile, then PV
    v16bf pfrag = frag_a(&plds[w][0][0], 32, lane);
#pragma unroll
    for (int nt = 0; nt < 8; ++nt) {
      v16bf vf = frag_b(vbt + ((size_t)kvh * HD + nt * 16) * S_LEN + kbase, S_LEN, lane);
      o[nt] = wmma_bf16(pfrag, vf, o[nt]);
    }
  }

#pragma unroll
  for (int nt = 0; nt < 8; ++nt)
#pragma unroll
    for (int r = 0; r < 8; ++r) {
      const float val = o[nt][r] / l[r];
      const int row = q0 + r + 8 * half;
      attn[(size_t)row * (NH * HD) + head * HD + nt * 16 + nn] = f2bf(val);
    }
}

// ---------------------------------------------------------------- launch
extern "C" void kernel_launch(void* const* d_in, const int* in_sizes, int n_in,
                              void* d_out, int out_size, void* d_ws, size_t ws_size,
                              hipStream_t stream) {
  (void)in_sizes; (void)n_in; (void)out_size; (void)ws_size;
  const float* hs   = (const float*)d_in[0];
  const float* wq   = (const float*)d_in[1];
  const float* wk   = (const float*)d_in[2];
  const float* wv   = (const float*)d_in[3];
  const float* wo   = (const float*)d_in[4];
  const float* qnw  = (const float*)d_in[5];
  const float* knw  = (const float*)d_in[6];
  const float* cosf = (const float*)d_in[7];
  const float* sinf = (const float*)d_in[8];
  // d_in[9] = attention_mask: causal, applied analytically in flash_attn.

  char* p = (char*)d_ws;
  auto alloc = [&](size_t bytes) -> char* {
    char* r = p;
    p += (bytes + 255) & ~(size_t)255;
    return r;
  };
  bf16_t* hs_bf = (bf16_t*)alloc((size_t)S_LEN * D_MODEL * 2);
  bf16_t* wq_bf = (bf16_t*)alloc((size_t)(NH * HD) * D_MODEL * 2);
  bf16_t* wk_bf = (bf16_t*)alloc((size_t)(NKV * HD) * D_MODEL * 2);
  bf16_t* wv_bf = (bf16_t*)alloc((size_t)(NKV * HD) * D_MODEL * 2);
  bf16_t* wo_bf = (bf16_t*)alloc((size_t)D_MODEL * (NH * HD) * 2);
  float*  q_f32 = (float*)alloc((size_t)S_LEN * (NH * HD) * 4);
  float*  k_f32 = (float*)alloc((size_t)S_LEN * (NKV * HD) * 4);
  float*  v_f32 = (float*)alloc((size_t)S_LEN * (NKV * HD) * 4);
  bf16_t* qb    = (bf16_t*)alloc((size_t)NH * S_LEN * HD * 2);
  bf16_t* kbuf  = (bf16_t*)alloc((size_t)NKV * S_LEN * HD * 2);
  bf16_t* vbt   = (bf16_t*)alloc((size_t)NKV * HD * S_LEN * 2);
  bf16_t* attn  = (bf16_t*)q_f32;  // q_f32 is dead once attn is written

  auto cvt = [&](const float* s, bf16_t* d, int n) {
    cvt_f32_bf16_kernel<<<(n + 255) / 256, 256, 0, stream>>>(s, d, n);
  };
  cvt(hs, hs_bf, S_LEN * D_MODEL);
  cvt(wq, wq_bf, NH * HD * D_MODEL);
  cvt(wk, wk_bf, NKV * HD * D_MODEL);
  cvt(wv, wv_bf, NKV * HD * D_MODEL);
  cvt(wo, wo_bf, D_MODEL * NH * HD);

  // projections: X @ W^T   (all GEMMs in this block have K = 4096)
  gemm_bf16_kernel<D_MODEL><<<dim3((NH * HD) / 128, S_LEN / 128), 256, 0, stream>>>(
      hs_bf, wq_bf, q_f32, S_LEN, NH * HD);
  gemm_bf16_kernel<D_MODEL><<<dim3((NKV * HD) / 128, S_LEN / 128), 256, 0, stream>>>(
      hs_bf, wk_bf, k_f32, S_LEN, NKV * HD);
  gemm_bf16_kernel<D_MODEL><<<dim3((NKV * HD) / 128, S_LEN / 128), 256, 0, stream>>>(
      hs_bf, wv_bf, v_f32, S_LEN, NKV * HD);

  // RMSNorm + RoPE + per-head bf16 layouts
  {
    const long rows = (long)S_LEN * NH + 2L * S_LEN * NKV;
    qkv_post_kernel<<<(int)((rows + 7) / 8), 256, 0, stream>>>(
        q_f32, k_f32, v_f32, qnw, knw, cosf, sinf, qb, kbuf, vbt);
  }

  // causal GQA attention (flash, online softmax)
  flash_attn_kernel<<<NH * (S_LEN / 128), 256, 0, stream>>>(qb, kbuf, vbt, attn);

  // output projection -> f32 d_out
  gemm_bf16_kernel<NH * HD><<<dim3(D_MODEL / 128, S_LEN / 128), 256, 0, stream>>>(
      attn, wo_bf, (float*)d_out, S_LEN, D_MODEL);
}
